// TTRBF_17583596110581
// MI455X (gfx1250) — compile-verified, matrix operand order checked by reading, not run on hardware
//
#include <hip/hip_runtime.h>

// ---------------------------------------------------------------------------
// MI455X (gfx1250) tensor-train RBF, bf16 WMMA formulation, 4 waves/tile.
//   phi  = exp(-max(||x||^2+||c||^2-2 x.c, 0) * exp(-2*log_sigma))
//   R0   = Phi0 @ G0                                   (WMMA, K=512)
//   t1[b,j] = sum_d Phi1[b,d] * (R0[b,:] . G1[:,d,j])  (WMMA K=64 per d)
//   t2, out analogous with G2, G3.
// One workgroup (4 wave32) per 16-row batch tile; the d/K loops are split
// across the 4 waves and reduced with ds_add_f32 in LDS. 256 blocks x 4
// waves = 1024 waves => ~2 waves per SIMD32 on a 128-WGP part.
// ---------------------------------------------------------------------------

typedef __attribute__((ext_vector_type(16))) __bf16    v16bf;
typedef __attribute__((ext_vector_type(8)))  float     v8f;
typedef __attribute__((ext_vector_type(4)))  unsigned  u32x4;

#define DEV static __device__ __forceinline__

DEV unsigned short f2bf_bits(float f) {
  unsigned u = __builtin_bit_cast(unsigned, f);
  unsigned r = (u + 0x7FFFu + ((u >> 16) & 1u)) >> 16;   // round-to-nearest-even
  return (unsigned short)r;
}
DEV __bf16 f2bf(float f) { return __builtin_bit_cast(__bf16, f2bf_bits(f)); }
DEV float bf2f(__bf16 b) {
  unsigned u = ((unsigned)__builtin_bit_cast(unsigned short, b)) << 16;
  return __builtin_bit_cast(float, u);
}
// 16-bit A/B fragment element -> K index (ISA 7.12.2, 16-bit 16x32 layout)
DEV int k_of(int e, int h) { return ((e >> 3) << 4) + (h << 3) + (e & 7); }

DEV v8f vzero8() { v8f z = {0.f,0.f,0.f,0.f,0.f,0.f,0.f,0.f}; return z; }

DEV v16bf load_a_lds(const __bf16* base, int ld, int m, int h, int k0) {
  // two contiguous 16B chunks per lane: [m][k0+h*8 .. +7] and [m][k0+16+h*8 .. +7]
  const __bf16* p = base + m * ld + k0 + h * 8;
  union { u32x4 q[2]; v16bf v; } u;
  u.q[0] = *reinterpret_cast<const u32x4*>(p);
  u.q[1] = *reinterpret_cast<const u32x4*>(p + 16);
  return u.v;
}

DEV void store_frag16(unsigned short* dst, const unsigned short* vals) {
  union { unsigned short s[16]; u32x4 q[2]; } u;
#pragma unroll
  for (int i = 0; i < 16; ++i) u.s[i] = vals[i];
  u32x4* d = reinterpret_cast<u32x4*>(dst);
  d[0] = u.q[0];
  d[1] = u.q[1];
}

// ------------------------------- prep kernels ------------------------------

__global__ void ttrbf_prep_stats(const float* __restrict__ centres,
                                 const float* __restrict__ log_sigmas,
                                 float* __restrict__ cn, float* __restrict__ sx) {
  int d = blockIdx.x * blockDim.x + threadIdx.x;
  if (d >= 2048) return;
  float s = 0.f;
#pragma unroll
  for (int k = 0; k < 32; ++k) { float v = centres[d * 32 + k]; s += v * v; }
  cn[d] = s;
  sx[d] = __expf(-2.0f * log_sigmas[d]);
}

__global__ void ttrbf_prep_centres(const float* __restrict__ centres,
                                   unsigned short* __restrict__ CB) {
  int t = blockIdx.x * blockDim.x + threadIdx.x;  // 128 tiles * 32 lanes
  if (t >= 128 * 32) return;
  int tile = t >> 5, lane = t & 31;
  int n = lane & 15, h = lane >> 4;
  unsigned short v[16];
#pragma unroll
  for (int e = 0; e < 16; ++e)
    v[e] = f2bf_bits(centres[(tile * 16 + n) * 32 + k_of(e, h)]);
  store_frag16(CB + (size_t)t * 16, v);
}

__global__ void ttrbf_prep_g0(const float* __restrict__ G0,
                              unsigned short* __restrict__ G0B) {
  int t = blockIdx.x * blockDim.x + threadIdx.x;  // 64 frags * 32 lanes
  if (t >= 64 * 32) return;
  int f = t >> 5, lane = t & 31;
  int kt = f >> 2, nt = f & 3;
  int n = lane & 15, h = lane >> 4;
  unsigned short v[16];
#pragma unroll
  for (int e = 0; e < 16; ++e) {
    int k = kt * 32 + k_of(e, h);
    v[e] = f2bf_bits(G0[k * 64 + nt * 16 + n]);
  }
  store_frag16(G0B + (size_t)t * 16, v);
}

// G layout (64, 512, ncols); fragment index fi = (d*NT + nt)*2 + kc
__global__ void ttrbf_prep_g(const float* __restrict__ G,
                             unsigned short* __restrict__ GB,
                             int ncols, int NT) {
  int t = blockIdx.x * blockDim.x + threadIdx.x;
  int total = 512 * NT * 2 * 32;
  if (t >= total) return;
  int lane = t & 31;
  int f = t >> 5;
  int kc = f & 1;
  int q = f >> 1;
  int nt = q % NT;
  int d  = q / NT;
  int n = lane & 15, h = lane >> 4;
  unsigned short v[16];
#pragma unroll
  for (int e = 0; e < 16; ++e) {
    int i = kc * 32 + k_of(e, h);
    v[e] = f2bf_bits(G[((size_t)i * 512 + d) * ncols + nt * 16 + n]);
  }
  store_frag16(GB + (size_t)t * 16, v);
}

// ------------------------------- main kernel -------------------------------

__global__ __launch_bounds__(128) void ttrbf_main(
    const float* __restrict__ X,
    const float* __restrict__ cn, const float* __restrict__ sx,
    const unsigned short* __restrict__ CB,
    const unsigned short* __restrict__ G0B,
    const unsigned short* __restrict__ G1B,
    const unsigned short* __restrict__ G2B,
    const unsigned short* __restrict__ G3B,
    float* __restrict__ out) {
  __shared__ __align__(16) __bf16 phi[16][512];   // current 16x512 phi chunk
  __shared__ __align__(16) __bf16 rbuf[16][64];   // R0 / t1 / t2 (bf16)
  __shared__ float facc[16][64];                  // f32 cross-wave reduction
  __shared__ float xnorm[16];

  const int tid  = threadIdx.x;
  const int lane = tid & 31;
  const int wv   = tid >> 5;                      // wave id 0..3
  const int m = lane & 15, h = lane >> 4;
  const int b0 = blockIdx.x * 16;

  if (tid < 16) {
    float s = 0.f;
#pragma unroll
    for (int k = 0; k < 32; ++k) { float v = X[(b0 + tid) * 32 + k]; s += v * v; }
    xnorm[tid] = s;
  }
  // A-fragment of the X row tile (16x32, K=32 exactly one WMMA per c-tile)
  v16bf xa;
#pragma unroll
  for (int e = 0; e < 16; ++e) xa[e] = f2bf(X[(b0 + m) * 32 + k_of(e, h)]);
  __syncthreads();

  for (int p = 0; p < 4; ++p) {
    // zero the f32 reduction buffer (safe: previous phase consumed it
    // before its trailing barrier)
    for (int i = tid; i < 16 * 64; i += 128) (&facc[0][0])[i] = 0.f;

    // ---- build phi chunk for this phase: wave wv does 8 of 32 c-tiles ----
    for (int dtl = 0; dtl < 8; ++dtl) {
      int dt = wv * 8 + dtl;
      int gt = p * 32 + dt;  // global 16-wide centre tile
      const v16bf cb = *reinterpret_cast<const v16bf*>(CB + (size_t)gt * 512 + lane * 16);
      v8f dot = __builtin_amdgcn_wmma_f32_16x16x32_bf16(
          false, xa, false, cb, (short)0, vzero8(), false, false);
      int gd = gt * 16 + m;              // this lane's centre column
      float cnv = cn[gd], sv = sx[gd];
#pragma unroll
      for (int r = 0; r < 8; ++r) {
        int row = h * 8 + r;
        float sq = xnorm[row] + cnv - 2.0f * dot[r];
        sq = fmaxf(sq, 0.0f);
        phi[row][dt * 16 + m] = f2bf(__expf(-sq * sv));
      }
    }
    __syncthreads();

    if (p == 0) {
      // R0 = Phi0 @ G0 : K=512 split 4 ways (wave wv -> kt = wv*4..wv*4+3)
      v8f acc[4];
#pragma unroll
      for (int nt = 0; nt < 4; ++nt) acc[nt] = vzero8();
      for (int ktl = 0; ktl < 4; ++ktl) {
        int kt = wv * 4 + ktl;
        v16bf a = load_a_lds(&phi[0][0], 512, m, h, kt * 32);
#pragma unroll
        for (int nt = 0; nt < 4; ++nt) {
          const v16bf b = *reinterpret_cast<const v16bf*>(
              G0B + (size_t)(kt * 4 + nt) * 512 + lane * 16);
          acc[nt] = __builtin_amdgcn_wmma_f32_16x16x32_bf16(
              false, a, false, b, (short)0, acc[nt], false, false);
        }
      }
#pragma unroll
      for (int nt = 0; nt < 4; ++nt)
#pragma unroll
        for (int r = 0; r < 8; ++r)
          atomicAdd(&facc[h * 8 + r][nt * 16 + m], acc[nt][r]);   // ds_add_f32
      __syncthreads();
      for (int i = tid; i < 16 * 64; i += 128)
        (&rbuf[0][0])[i] = f2bf((&facc[0][0])[i]);
      __syncthreads();
    } else {
      const int NT = (p == 3) ? 2 : 4;   // output cols: 64,64,32
      const unsigned short* GB = (p == 1) ? G1B : ((p == 2) ? G2B : G3B);
      // A-fragments of the previous stage result (16x64, K=64 -> 2 fragments)
      v16bf a0 = load_a_lds(&rbuf[0][0], 64, m, h, 0);
      v16bf a1 = load_a_lds(&rbuf[0][0], 64, m, h, 32);
      float acc[4][8];
#pragma unroll
      for (int nt = 0; nt < 4; ++nt)
#pragma unroll
        for (int r = 0; r < 8; ++r) acc[nt][r] = 0.f;

      const size_t dstride = (size_t)NT * 2 * 512;  // ushorts per d
      // wave wv handles d in [wv*128, wv*128+128)
      for (int dl = 0; dl < 128; ++dl) {
        int d = wv * 128 + dl;
        const unsigned short* fb = GB + (size_t)d * dstride;
        __builtin_prefetch(fb + dstride, 0, 1);     // global_prefetch next d
        float pw[8];
#pragma unroll
        for (int r = 0; r < 8; ++r) pw[r] = bf2f(phi[h * 8 + r][d]);
        for (int nt = 0; nt < NT; ++nt) {
          const v16bf bv0 = *reinterpret_cast<const v16bf*>(
              fb + (size_t)(nt * 2 + 0) * 512 + lane * 16);
          const v16bf bv1 = *reinterpret_cast<const v16bf*>(
              fb + (size_t)(nt * 2 + 1) * 512 + lane * 16);
          v8f v = __builtin_amdgcn_wmma_f32_16x16x32_bf16(
              false, a0, false, bv0, (short)0, vzero8(), false, false);
          v = __builtin_amdgcn_wmma_f32_16x16x32_bf16(
              false, a1, false, bv1, (short)0, v, false, false);
#pragma unroll
          for (int r = 0; r < 8; ++r) acc[nt][r] = fmaf(pw[r], v[r], acc[nt][r]);
        }
      }
#pragma unroll
      for (int nt = 0; nt < 4; ++nt) {
        if (nt >= NT) break;
#pragma unroll
        for (int r = 0; r < 8; ++r)
          atomicAdd(&facc[h * 8 + r][nt * 16 + m], acc[nt][r]);   // ds_add_f32
      }
      __syncthreads();
      if (p < 3) {
        for (int i = tid; i < 16 * 64; i += 128)
          (&rbuf[0][0])[i] = f2bf((&facc[0][0])[i]);
        __syncthreads();
      } else {
        for (int i = tid; i < 16 * 32; i += 128) {
          int row = i >> 5, col = i & 31;
          out[(size_t)(b0 + row) * 32 + col] = facc[row][col];
        }
      }
    }
  }
}

// ------------------------------- launcher ----------------------------------

extern "C" void kernel_launch(void* const* d_in, const int* in_sizes, int n_in,
                              void* d_out, int out_size, void* d_ws, size_t ws_size,
                              hipStream_t stream) {
  const float* X          = (const float*)d_in[0];
  const float* centres    = (const float*)d_in[1];
  const float* log_sigmas = (const float*)d_in[2];
  const float* G0         = (const float*)d_in[3];
  const float* G1         = (const float*)d_in[4];
  const float* G2         = (const float*)d_in[5];
  const float* G3         = (const float*)d_in[6];

  char* ws = (char*)d_ws;
  float*          cn  = (float*)(ws + 0);                 //   8 KB
  float*          sx  = (float*)(ws + 8192);              //   8 KB
  unsigned short* CB  = (unsigned short*)(ws + 16384);    // 128 KB
  size_t off = 16384 + 131072;
  unsigned short* G0B = (unsigned short*)(ws + off); off += 65536;        //  64 KB
  unsigned short* G1B = (unsigned short*)(ws + off); off += 4u*1024*1024; //   4 MB
  unsigned short* G2B = (unsigned short*)(ws + off); off += 4u*1024*1024; //   4 MB
  unsigned short* G3B = (unsigned short*)(ws + off);                      //   2 MB

  ttrbf_prep_stats  <<<8,   256, 0, stream>>>(centres, log_sigmas, cn, sx);
  ttrbf_prep_centres<<<16,  256, 0, stream>>>(centres, CB);
  ttrbf_prep_g0     <<<8,   256, 0, stream>>>(G0, G0B);
  ttrbf_prep_g      <<<512, 256, 0, stream>>>(G1, G1B, 64, 4);
  ttrbf_prep_g      <<<512, 256, 0, stream>>>(G2, G2B, 64, 4);
  ttrbf_prep_g      <<<256, 256, 0, stream>>>(G3, G3B, 32, 2);
  ttrbf_main        <<<256, 128, 0, stream>>>(X, cn, sx, CB, G0B, G1B, G2B, G3B,
                                              (float*)d_out);
}
